// GATOccupancyPredictor_89910845375334
// MI455X (gfx1250) — matrix operand order repaired and checked by reading, not compiled, not run on hardware
//
#include <hip/hip_runtime.h>
#include <hip/hip_bf16.h>

// ---------------- problem constants ----------------
#define BB 8
#define PP 2048
#define NPTS (BB * PP)      // 16384 nodes
#define KNN 16
#define HEADS 4
#define NEG_SLOPE 0.2f
#define BN_EPS 1e-5f

typedef __attribute__((ext_vector_type(16))) _Float16 v16h;
typedef __attribute__((ext_vector_type(8)))  _Float16 v8h;
typedef __attribute__((ext_vector_type(8)))  float    v8f;

// =====================================================================
// KNN: per batch, top-16 nearest neighbors (exclude self).
// One thread per query point; candidates staged through LDS in 256-chunks.
// Sorted top-K in registers; insert-at-end + one unrolled bubble pass
// (static indices -> no scratch spills).
// =====================================================================
__global__ __launch_bounds__(256) void knn_kernel(const float* __restrict__ pos,
                                                  int* __restrict__ srcIdx) {
    __shared__ float sx[256], sy[256], sz[256];
    const int b = blockIdx.y;
    const int q = blockIdx.x * 256 + threadIdx.x;
    const float* px = pos + (size_t)(b * 3 + 0) * PP;
    const float* py = pos + (size_t)(b * 3 + 1) * PP;
    const float* pz = pos + (size_t)(b * 3 + 2) * PP;
    const float qx = px[q], qy = py[q], qz = pz[q];

    float bd[KNN];
    int   bi[KNN];
#pragma unroll
    for (int i = 0; i < KNN; ++i) { bd[i] = 3.4e38f; bi[i] = 0; }

    for (int c0 = 0; c0 < PP; c0 += 256) {
        __syncthreads();
        sx[threadIdx.x] = px[c0 + threadIdx.x];
        sy[threadIdx.x] = py[c0 + threadIdx.x];
        sz[threadIdx.x] = pz[c0 + threadIdx.x];
        __syncthreads();
        for (int j = 0; j < 256; ++j) {
            const int cand = c0 + j;
            const float dx = qx - sx[j], dy = qy - sy[j], dz = qz - sz[j];
            float d2 = dx * dx + dy * dy + dz * dz;
            if (cand == q) d2 = 3.9e38f;
            if (d2 < bd[KNN - 1]) {
                bd[KNN - 1] = d2; bi[KNN - 1] = cand;
#pragma unroll
                for (int t = KNN - 1; t > 0; --t) {
                    if (bd[t] < bd[t - 1]) {
                        float td = bd[t]; bd[t] = bd[t - 1]; bd[t - 1] = td;
                        int   ti = bi[t]; bi[t] = bi[t - 1]; bi[t - 1] = ti;
                    }
                }
            }
        }
    }
    const int n = b * PP + q;
#pragma unroll
    for (int k = 0; k < KNN; ++k) srcIdx[n * KNN + k] = b * PP + bi[k];
}

// =====================================================================
// WMMA GEMM (f16 in, f32 acc): Y[N,Cout] = Xh[N,Cin] @ Wh[Cout,Cin]^T (+bias)
// One wave per 16(M) x 64(N) tile: shared A fragment, 4 WMMAs per K-step.
// A/B lane layouts: contiguous K, k = 16*(lane/16) + s, same for A and B (16-bit 16x32 A, 32x16 B).
// =====================================================================
__global__ __launch_bounds__(32) void gemm_wmma_kernel(const _Float16* __restrict__ Xh,
                                                       const _Float16* __restrict__ Wh,
                                                       const float* __restrict__ bias,
                                                       float* __restrict__ Y,
                                                       int Cin, int Cout) {
    const int lane  = threadIdx.x;
    const int l15   = lane & 15;
    const bool hi   = lane >= 16;
    const int mbase = blockIdx.x * 16;
    const int nbase = blockIdx.y * 64;

    const _Float16* xr = Xh + (size_t)(mbase + l15) * Cin;   // A row (M = l15)
    const _Float16* wr0 = Wh + (size_t)(nbase +  0 + l15) * Cin;
    const _Float16* wr1 = Wh + (size_t)(nbase + 16 + l15) * Cin;
    const _Float16* wr2 = Wh + (size_t)(nbase + 32 + l15) * Cin;
    const _Float16* wr3 = Wh + (size_t)(nbase + 48 + l15) * Cin;

    v8f c0 = {}, c1 = {}, c2 = {}, c3 = {};
    for (int k0 = 0; k0 < Cin; k0 += 32) {
        // A 16x32 f16: lanes<16 -> K = k0+[0..15]; lanes>=16 -> K = k0+16+[0..15]
        const int aLo = k0 + (hi ? 16 : 0);
        v16h a = *(const v16h*)(xr + aLo);

        // B 32x16 f16: lanes<16 -> K = k0+[0..15]; lanes>=16 -> K = k0+16+[0..15]
        const int bK = k0 + (hi ? 16 : 0);
        v16h b0 = *(const v16h*)(wr0 + bK);
        v16h b1 = *(const v16h*)(wr1 + bK);
        v16h b2 = *(const v16h*)(wr2 + bK);
        v16h b3 = *(const v16h*)(wr3 + bK);

        if (k0 + 32 < Cin) __builtin_prefetch(xr + k0 + 32, 0, 1);

        c0 = __builtin_amdgcn_wmma_f32_16x16x32_f16(false, a, false, b0, (short)0, c0, false, false);
        c1 = __builtin_amdgcn_wmma_f32_16x16x32_f16(false, a, false, b1, (short)0, c1, false, false);
        c2 = __builtin_amdgcn_wmma_f32_16x16x32_f16(false, a, false, b2, (short)0, c2, false, false);
        c3 = __builtin_amdgcn_wmma_f32_16x16x32_f16(false, a, false, b3, (short)0, c3, false, false);
    }

#pragma unroll
    for (int t = 0; t < 4; ++t) {
        const v8f c = (t == 0) ? c0 : (t == 1) ? c1 : (t == 2) ? c2 : c3;
        const int col = nbase + 16 * t + l15;
        const float bv = bias ? bias[col] : 0.0f;
#pragma unroll
        for (int r = 0; r < 8; ++r) {          // D layout: element r -> M = r (+8 for hi half)
            const int m = mbase + r + (hi ? 8 : 0);
            Y[(size_t)m * Cout + col] = c[r] + bv;
        }
    }
}

// =====================================================================
// f32 -> f16 conversion (weights, once per launch)
// =====================================================================
__global__ __launch_bounds__(256) void f32tof16_kernel(const float* __restrict__ x,
                                                       _Float16* __restrict__ y, int n) {
    const int i = blockIdx.x * 256 + threadIdx.x;
    if (i < n) y[i] = (_Float16)x[i];
}

// =====================================================================
// in1: [N,3] @ W[64,3]^T + b  (K=3); reads pos [B,3,P] directly
// =====================================================================
__global__ __launch_bounds__(256) void in1_kernel(const float* __restrict__ pos,
                                                  const float* __restrict__ w,
                                                  const float* __restrict__ bias,
                                                  float* __restrict__ y) {
    const int t = blockIdx.x * 256 + threadIdx.x;   // n*64 + co
    const int n = t >> 6, co = t & 63;
    if (n >= NPTS) return;
    const int b = n / PP, p = n % PP;
    const float x0 = pos[(size_t)(b * 3 + 0) * PP + p];
    const float x1 = pos[(size_t)(b * 3 + 1) * PP + p];
    const float x2 = pos[(size_t)(b * 3 + 2) * PP + p];
    y[t] = x0 * w[co * 3 + 0] + x1 * w[co * 3 + 1] + x2 * w[co * 3 + 2] + bias[co];
}

// =====================================================================
// BatchNorm (training-mode batch stats): accumulate -> finalize -> apply
// =====================================================================
__global__ __launch_bounds__(256) void colstat_accum_kernel(const float* __restrict__ X,
                                                            float* __restrict__ acc, int C) {
    __shared__ float s[1024];                 // [sum(C) | sumsq(C)]
    for (int i = threadIdx.x; i < 2 * C; i += 256) s[i] = 0.0f;
    __syncthreads();
    const int r0 = blockIdx.x * 64;
    for (int idx = threadIdx.x; idx < 64 * C; idx += 256) {
        const int r = r0 + idx / C, c = idx % C;
        const float v = X[(size_t)r * C + c];
        atomicAdd(&s[c], v);
        atomicAdd(&s[C + c], v * v);
    }
    __syncthreads();
    for (int c = threadIdx.x; c < C; c += 256) {
        atomicAdd(&acc[c], s[c]);
        atomicAdd(&acc[C + c], s[C + c]);
    }
}

__global__ __launch_bounds__(256) void bn_finalize_kernel(const float* __restrict__ acc,
                                                          const float* __restrict__ g,
                                                          const float* __restrict__ b,
                                                          float* __restrict__ scsh, int C) {
    const int c = blockIdx.x * 256 + threadIdx.x;
    if (c >= C) return;
    const float m  = acc[c] * (1.0f / NPTS);
    const float v  = acc[C + c] * (1.0f / NPTS) - m * m;   // biased variance
    const float rs = rsqrtf(v + BN_EPS);
    const float sc = g[c] * rs;
    scsh[c]     = sc;
    scsh[C + c] = b[c] - m * sc;
}

// out = relu?( x*scale + shift ) (+ residual added AFTER relu, as in reference)
// Optionally dual-writes an f16 copy for downstream WMMA GEMMs.
__global__ __launch_bounds__(256) void bn_apply_kernel(const float* __restrict__ X,
                                                       const float* __restrict__ scsh,
                                                       const float* __restrict__ res,
                                                       float* __restrict__ Y,
                                                       _Float16* __restrict__ Y16,
                                                       int C, int relu) {
    const size_t i = (size_t)blockIdx.x * 256 + threadIdx.x;
    if (i >= (size_t)NPTS * C) return;
    const int c = (int)(i % C);
    float v = X[i] * scsh[c] + scsh[C + c];
    if (relu) v = fmaxf(v, 0.0f);
    if (res)  v += res[i];
    Y[i] = v;
    if (Y16) Y16[i] = (_Float16)v;
}

// =====================================================================
// GAT: per-node attention logits s,d ; softmax over K+1 edges + gather
// =====================================================================
__global__ __launch_bounds__(256) void gat_logits_kernel(const float* __restrict__ h,
                                                         const float* __restrict__ a_src,
                                                         const float* __restrict__ a_dst,
                                                         float* __restrict__ sd, int oc) {
    const int t = blockIdx.x * 256 + threadIdx.x;   // n*HEADS + hd
    const int n = t >> 2, hd = t & 3;
    if (n >= NPTS) return;
    const float* hr = h + (size_t)n * HEADS * oc + (size_t)hd * oc;
    const float* as = a_src + hd * oc;
    const float* ad = a_dst + hd * oc;
    float s = 0.0f, d = 0.0f;
    for (int c = 0; c < oc; ++c) { const float v = hr[c]; s += v * as[c]; d += v * ad[c]; }
    sd[n * 8 + hd]     = s;
    sd[n * 8 + 4 + hd] = d;
}

__global__ __launch_bounds__(128) void gat_aggregate_kernel(const float* __restrict__ h,
                                                            const float* __restrict__ sd,
                                                            const int* __restrict__ srcIdx,
                                                            const float* __restrict__ bias,
                                                            float* __restrict__ out, int oc) {
    const int n = blockIdx.x;
    const int C = HEADS * oc;
    __shared__ float al[17 * 4];
    __shared__ int   sj[17];
    const int tid = threadIdx.x;

    if (tid < 17) {                           // 16 knn edges + self loop
        const int s = (tid < KNN) ? srcIdx[n * KNN + tid] : n;
        sj[tid] = s;
#pragma unroll
        for (int hd = 0; hd < HEADS; ++hd) {
            const float e = sd[s * 8 + hd] + sd[n * 8 + 4 + hd];
            al[tid * 4 + hd] = (e > 0.0f) ? e : NEG_SLOPE * e;
        }
    }
    __syncthreads();
    if (tid < HEADS) {                        // segment softmax, 17 logits per head
        float mx = -3.4e38f;
#pragma unroll
        for (int j = 0; j < 17; ++j) mx = fmaxf(mx, al[j * 4 + tid]);
        float ex[17], den = 0.0f;
#pragma unroll
        for (int j = 0; j < 17; ++j) { ex[j] = __expf(al[j * 4 + tid] - mx); den += ex[j]; }
        const float inv = 1.0f / (den + 1e-16f);
#pragma unroll
        for (int j = 0; j < 17; ++j) al[j * 4 + tid] = ex[j] * inv;
    }
    __syncthreads();
    for (int c = tid; c < C; c += 128) {      // weighted gather (L2-resident h)
        const int hd = c / oc;
        float acc = 0.0f;
#pragma unroll
        for (int j = 0; j < 17; ++j) acc += al[j * 4 + hd] * h[(size_t)sj[j] * C + c];
        out[(size_t)n * C + c] = acc + bias[c];
    }
}

// =====================================================================
// concat [y256 | x1] -> 512 (f16, feeds GEMM only); final 128->2 + transpose
// =====================================================================
__global__ __launch_bounds__(256) void concat_kernel(const float* __restrict__ a,
                                                     const float* __restrict__ b,
                                                     _Float16* __restrict__ y) {
    const size_t i = (size_t)blockIdx.x * 256 + threadIdx.x;
    if (i >= (size_t)NPTS * 512) return;
    const int c = (int)(i & 511);
    const size_t n = i >> 9;
    const float v = (c < 256) ? a[n * 256 + c] : b[n * 256 + (c - 256)];
    y[i] = (_Float16)v;
}

__global__ __launch_bounds__(256) void fi2_kernel(const float* __restrict__ x,
                                                  const float* __restrict__ w,
                                                  const float* __restrict__ bias,
                                                  float* __restrict__ out) {
    const int n = blockIdx.x * 256 + threadIdx.x;
    if (n >= NPTS) return;
    const float* xr = x + (size_t)n * 128;
    float a0 = bias[0], a1 = bias[1];
    for (int k = 0; k < 128; ++k) { const float v = xr[k]; a0 += v * w[k]; a1 += v * w[128 + k]; }
    const int b = n / PP, p = n % PP;
    out[(size_t)(b * 2 + 0) * PP + p] = a0;    // [B, 2, P]
    out[(size_t)(b * 2 + 1) * PP + p] = a1;
}

// =====================================================================
// host orchestration
// =====================================================================
extern "C" void kernel_launch(void* const* d_in, const int* in_sizes, int n_in,
                              void* d_out, int out_size, void* d_ws, size_t ws_size,
                              hipStream_t stream) {
    (void)in_sizes; (void)out_size; (void)ws_size;
    const float* F[64];
    for (int i = 0; i < 57 && i < n_in; ++i) F[i] = (const float*)d_in[i];
    const float* pos = F[0];
    float* out = (float*)d_out;

    // ---- workspace bump allocator ----
    size_t off = 0;
    auto alloc = [&](size_t bytes) -> void* {
        void* p = (char*)d_ws + off;
        off += (bytes + 255) & ~(size_t)255;
        return p;
    };
    const size_t R = (size_t)NPTS;
    int*   src   = (int*)alloc(R * KNN * sizeof(int));
    float* sd    = (float*)alloc(R * 8 * sizeof(float));
    float* acc   = (float*)alloc(2 * 512 * sizeof(float));
    float* scsh  = (float*)alloc(2 * 512 * sizeof(float));
    float* t64a  = (float*)alloc(R * 64 * sizeof(float));
    float* t64b  = (float*)alloc(R * 64 * sizeof(float));
    float* s256[4]; for (int i = 0; i < 4; ++i) s256[i] = (float*)alloc(R * 256 * sizeof(float));
    float* s512[4]; for (int i = 0; i < 4; ++i) s512[i] = (float*)alloc(R * 512 * sizeof(float));
    _Float16* h64  = (_Float16*)alloc(R * 64 * sizeof(_Float16));   // f16 activation slots
    _Float16* h256 = (_Float16*)alloc(R * 256 * sizeof(_Float16));  // (reused along the chain)
    _Float16* h512 = (_Float16*)alloc(R * 512 * sizeof(_Float16));
    _Float16* warena = (_Float16*)alloc(2 * 1024 * 1024 * sizeof(_Float16));

    // ---- convert GEMM weights to f16 once ----
    const int widx[11]  = {5, 9, 13, 19, 23, 29, 35, 39, 43, 47, 51};
    const int wsizes[11] = {64 * 64, 256 * 64, 256 * 64, 512 * 256, 512 * 256, 512 * 512,
                            512 * 512, 256 * 512, 256 * 512, 256 * 256, 128 * 256};
    _Float16* Wh[64];
    {
        size_t woff = 0;
        for (int i = 0; i < 11; ++i) {
            _Float16* dst = warena + woff;
            Wh[widx[i]] = dst;
            f32tof16_kernel<<<(wsizes[i] + 255) / 256, 256, 0, stream>>>(F[widx[i]], dst, wsizes[i]);
            woff += (size_t)wsizes[i];
        }
    }

    auto gemm = [&](const _Float16* Xh, int wi, const float* bias, float* Y,
                    int Cin, int Cout) {
        dim3 grid(NPTS / 16, Cout / 64);
        gemm_wmma_kernel<<<grid, 32, 0, stream>>>(Xh, Wh[wi], bias, Y, Cin, Cout);
    };
    auto bnstats = [&](const float* X, int C, const float* g, const float* b) {
        hipMemsetAsync(acc, 0, 2 * C * sizeof(float), stream);
        colstat_accum_kernel<<<NPTS / 64, 256, 0, stream>>>(X, acc, C);
        bn_finalize_kernel<<<(C + 255) / 256, 256, 0, stream>>>(acc, g, b, scsh, C);
    };
    auto bnapply = [&](const float* X, const float* res, float* Y, _Float16* Y16,
                       int C, int relu) {
        const size_t n = R * (size_t)C;
        bn_apply_kernel<<<(int)((n + 255) / 256), 256, 0, stream>>>(X, scsh, res, Y, Y16, C, relu);
    };
    auto gat = [&](const float* h, const float* as, const float* ad, const float* bias,
                   float* outp, int oc) {
        gat_logits_kernel<<<(NPTS * HEADS + 255) / 256, 256, 0, stream>>>(h, as, ad, sd, oc);
        gat_aggregate_kernel<<<NPTS, 128, 0, stream>>>(h, sd, src, bias, outp, oc);
    };

    // ---- 0) KNN graph (no_grad path) ----
    knn_kernel<<<dim3(PP / 256, BB), 256, 0, stream>>>(pos, src);

    // ---- 1) input MLP: 3 -> 64 -> 64 ----
    in1_kernel<<<(NPTS * 64 + 255) / 256, 256, 0, stream>>>(pos, F[1], F[2], t64a);
    bnstats(t64a, 64, F[3], F[4]);  bnapply(t64a, nullptr, t64b, h64, 64, 1);
    gemm(h64, 5, F[6], t64a, 64, 64);
    bnstats(t64a, 64, F[7], F[8]);  bnapply(t64a, nullptr, t64b, h64, 64, 1);   // x

    // ---- 2) block 1: sc1 + gat1 ----
    gemm(h64, 9, F[10], s256[0], 64, 256);
    bnstats(s256[0], 256, F[11], F[12]); bnapply(s256[0], nullptr, s256[1], nullptr, 256, 0); // id1
    gemm(h64, 13, nullptr, s256[0], 64, 256);                                                // h1
    gat(s256[0], F[14], F[15], F[16], s256[2], 64);                                          // agg1
    bnstats(s256[2], 256, F[17], F[18]); bnapply(s256[2], s256[1], s256[3], h256, 256, 1);    // x1

    // ---- 3) block 2: sc2 + gat2 ----
    gemm(h256, 19, F[20], s512[0], 256, 512);
    bnstats(s512[0], 512, F[21], F[22]); bnapply(s512[0], nullptr, s512[1], nullptr, 512, 0); // id2
    gemm(h256, 23, nullptr, s512[0], 256, 512);                                               // h2
    gat(s512[0], F[24], F[25], F[26], s512[2], 128);                                          // agg2
    bnstats(s512[2], 512, F[27], F[28]); bnapply(s512[2], nullptr, s512[3], h512, 512, 1);    // x2

    // ---- 4) block 3: gat3 (+ id2 residual) ----
    gemm(h512, 29, nullptr, s512[0], 512, 512);                                               // h3
    gat(s512[0], F[30], F[31], F[32], s512[2], 128);                                          // agg3
    bnstats(s512[2], 512, F[33], F[34]); bnapply(s512[2], s512[1], s512[0], h512, 512, 1);    // x3

    // ---- 5) decoder ----
    gemm(h512, 35, F[36], s512[1], 512, 512);
    bnstats(s512[1], 512, F[37], F[38]); bnapply(s512[1], nullptr, s512[2], h512, 512, 1);    // y512
    gemm(h512, 39, F[40], s256[0], 512, 256);
    bnstats(s256[0], 256, F[41], F[42]); bnapply(s256[0], nullptr, s256[1], nullptr, 256, 1); // y256

    // ---- 6) fuse: concat(y256, x1) -> fu1 -> fu2 ----
    concat_kernel<<<(int)((R * 512 + 255) / 256), 256, 0, stream>>>(s256[1], s256[3], h512);
    gemm(h512, 43, F[44], s256[0], 512, 256);
    bnstats(s256[0], 256, F[45], F[46]); bnapply(s256[0], nullptr, s256[2], h256, 256, 1);
    gemm(h256, 47, F[48], s256[0], 256, 256);
    bnstats(s256[0], 256, F[49], F[50]); bnapply(s256[0], nullptr, s256[1], h256, 256, 1);

    // ---- 7) final: 256 -> 128 -> 2 (+ output transpose) ----
    gemm(h256, 51, F[52], s256[0], 256, 128);
    bnstats(s256[0], 128, F[53], F[54]); bnapply(s256[0], nullptr, s256[2], nullptr, 128, 1);
    fi2_kernel<<<(NPTS + 255) / 256, 256, 0, stream>>>(s256[2], F[55], F[56], out);
}
